// SDCN_Spatial_20950850470045
// MI455X (gfx1250) — matrix operand order repaired and checked by reading, not compiled, not run on hardware
//
#include <hip/hip_runtime.h>
#include <math.h>
#include <stdint.h>

// ---------------------------------------------------------------------------
// Types for CDNA5 WMMA (gfx1250, wave32)
// ---------------------------------------------------------------------------
typedef __attribute__((ext_vector_type(16))) __bf16 v16bf;
typedef __attribute__((ext_vector_type(8)))  __bf16 v8bf;
typedef __attribute__((ext_vector_type(8)))  float  v8f;

union BFrag16 { v16bf v; struct { v8bf lo; v8bf hi; } h; };

static constexpr int kN      = 10000;   // nodes
static constexpr int kE      = 160000;  // edges
static constexpr int kHeads  = 4;
static constexpr float kSigma = 0.5f;

// ---------------------------------------------------------------------------
// GEMM: out[M,N] = act( A_bf16[M,K] @ B_bf16[K,N] + bias[N] )
// Workgroup tile 128x64, K-step 32. 8 waves; each wave does a 32x32 patch as
// 2x2 v_wmma_f32_16x16x32_bf16.
//
// Software-pipelined LDS staging with CDNA5 GLOBAL_LOAD_ASYNC_TO_LDS_B128
// (per-lane 16B memory->LDS DMA, ASYNCcnt-tracked): double-buffered tiles,
// loads for step s+1 are issued before computing step s, then a single
// s_wait_asynccnt 0 + workgroup barrier per K-step. K and N are multiples of
// 8 here, so every 8-elem chunk is fully in range or fully out -> OOB chunks
// are a plain zero ds_store.
//
// Fragment layouts per ISA 7.12.2:
//   A frag : lane(0..15)=row M, lane>>4 selects K-half {0..7 | 8..15}, then +16
//   B frag : lane&15 = col N, lane>>4 selects K-half {0..7 | 8..15}, then +16
//   C/D    : lane&15 = N, lane>>4 selects M rows {0..7 | 8..15}, 8 VGPRs
// ---------------------------------------------------------------------------
__global__ __launch_bounds__(256)
void gemm_bf16_wmma(const __bf16* __restrict__ A, int lda,
                    const __bf16* __restrict__ B, int ldb,
                    const float* __restrict__ bias,
                    float* __restrict__ C, int ldc,
                    int M, int N, int K, int act)
{
    constexpr int TM = 128, TN = 64, TK = 32;
    __shared__ __align__(16) __bf16 As[2][TM * TK];   // 2 x 8 KB
    __shared__ __align__(16) __bf16 Bs[2][TK * TN];   // 2 x 4 KB

    const int tid  = threadIdx.x;
    const int lane = tid & 31;
    const int wave = tid >> 5;
    const int wr   = wave & 3;   // wave row -> M offset wr*32
    const int wc   = wave >> 2;  // wave col -> N offset wc*32
    const int m15  = lane & 15;
    const int half = lane >> 4;

    const int blockM = blockIdx.y * TM;
    const int blockN = blockIdx.x * TN;

    const v8bf bzero = { (__bf16)0.0f, (__bf16)0.0f, (__bf16)0.0f, (__bf16)0.0f,
                         (__bf16)0.0f, (__bf16)0.0f, (__bf16)0.0f, (__bf16)0.0f };

    v8f acc[2][2];
    const v8f vzero = {0.f, 0.f, 0.f, 0.f, 0.f, 0.f, 0.f, 0.f};
    for (int i = 0; i < 2; ++i)
        for (int j = 0; j < 2; ++j) acc[i][j] = vzero;

    // Per-thread staging coordinates (chunk = 8 bf16 = 16 bytes).
    // A tile: 128x32 -> 512 chunks, 2 per thread. B tile: 32x64 -> 256 chunks.
    const int arow0 = tid >> 2, acol = (tid & 3) * 8;
    const int arow1 = arow0 + 64;                      // == (tid+256)>>2
    const int brow  = tid >> 3, bcol = (tid & 7) * 8;

    // Loop-invariant row/col validity + base global pointers.
    const bool aok0 = (blockM + arow0) < M;
    const bool aok1 = (blockM + arow1) < M;
    const bool bok  = (blockN + bcol) < N;
    const __bf16* Arow0 = A + (size_t)(blockM + arow0) * lda + acol;
    const __bf16* Arow1 = A + (size_t)(blockM + arow1) * lda + acol;
    const __bf16* Brow  = B + (size_t)brow * ldb + blockN + bcol;

    const uint32_t aoff0[2] = { (uint32_t)(uintptr_t)&As[0][arow0 * TK + acol],
                                (uint32_t)(uintptr_t)&As[1][arow0 * TK + acol] };
    const uint32_t aoff1[2] = { (uint32_t)(uintptr_t)&As[0][arow1 * TK + acol],
                                (uint32_t)(uintptr_t)&As[1][arow1 * TK + acol] };
    const uint32_t boff[2]  = { (uint32_t)(uintptr_t)&Bs[0][brow * TN + bcol],
                                (uint32_t)(uintptr_t)&Bs[1][brow * TN + bcol] };

    auto stage = [&](int k0, int buf) {
        if (aok0 && (k0 + acol) < K) {
            asm volatile("global_load_async_to_lds_b128 %0, %1, off"
                         :: "v"(aoff0[buf]), "v"(Arow0 + k0) : "memory");
        } else {
            *(v8bf*)(&As[buf][arow0 * TK + acol]) = bzero;
        }
        if (aok1 && (k0 + acol) < K) {
            asm volatile("global_load_async_to_lds_b128 %0, %1, off"
                         :: "v"(aoff1[buf]), "v"(Arow1 + k0) : "memory");
        } else {
            *(v8bf*)(&As[buf][arow1 * TK + acol]) = bzero;
        }
        if (bok && (k0 + brow) < K) {
            asm volatile("global_load_async_to_lds_b128 %0, %1, off"
                         :: "v"(boff[buf]), "v"(Brow + (size_t)k0 * ldb) : "memory");
        } else {
            *(v8bf*)(&Bs[buf][brow * TN + bcol]) = bzero;
        }
    };

    const int nsteps = (K + TK - 1) / TK;

    // Pipeline prologue: stage tile 0.
    stage(0, 0);
    asm volatile("s_wait_asynccnt 0x0" ::: "memory");
    __syncthreads();

    for (int s = 0; s < nsteps; ++s) {
        const int cur = s & 1;
        // Issue DMA for next tile while this tile computes.
        if (s + 1 < nsteps) stage((s + 1) * TK, cur ^ 1);

        const __bf16* Asb = &As[cur][0];
        const __bf16* Bsb = &Bs[cur][0];

        BFrag16 af[2], bfr[2];
        for (int mt = 0; mt < 2; ++mt) {
            int row = wr * 32 + mt * 16 + m15;
            af[mt].h.lo = *(const v8bf*)(Asb + row * TK + half * 8);        // K 0..7 / 8..15
            af[mt].h.hi = *(const v8bf*)(Asb + row * TK + 16 + half * 8);   // K 16..23 / 24..31
        }
        for (int nt = 0; nt < 2; ++nt) {
            int col = wc * 32 + nt * 16;
            for (int s = 0; s < 8; ++s) {
                bfr[nt].h.lo[s] = Bsb[(half * 8 + s) * TN + col + m15];        // K 0..7 / 8..15
                bfr[nt].h.hi[s] = Bsb[(16 + half * 8 + s) * TN + col + m15];   // K 16..23 / 24..31
            }
        }
        for (int mt = 0; mt < 2; ++mt)
            for (int nt = 0; nt < 2; ++nt)
                acc[mt][nt] = __builtin_amdgcn_wmma_f32_16x16x32_bf16(
                    false, af[mt].v, false, bfr[nt].v,
                    (short)0, acc[mt][nt], false, false);

        // Next tile's DMA must be complete; all waves done reading cur.
        asm volatile("s_wait_asynccnt 0x0" ::: "memory");
        __syncthreads();
    }

    // ---- epilogue: bias + optional ReLU (lane = column -> coalesced stores)
    for (int mt = 0; mt < 2; ++mt) {
        for (int nt = 0; nt < 2; ++nt) {
            int col = blockN + wc * 32 + nt * 16 + m15;
            if (col >= N) continue;
            float bc = bias ? bias[col] : 0.0f;
            for (int r = 0; r < 8; ++r) {
                int row = blockM + wr * 32 + mt * 16 + half * 8 + r;
                if (row < M) {
                    float v = acc[mt][nt][r] + bc;
                    if (act) v = v > 0.0f ? v : 0.0f;
                    C[(size_t)row * ldc + col] = v;
                }
            }
        }
    }
}

// ---------------------------------------------------------------------------
// Elementwise helpers
// ---------------------------------------------------------------------------
__global__ void cast_f32_bf16_k(const float* __restrict__ in, __bf16* __restrict__ out, size_t n) {
    size_t i = (size_t)blockIdx.x * blockDim.x + threadIdx.x;
    if (i < n) out[i] = (__bf16)in[i];
}
__global__ void fill_f32_k(float* p, float v, size_t n) {
    size_t i = (size_t)blockIdx.x * blockDim.x + threadIdx.x;
    if (i < n) p[i] = v;
}
__global__ void fill_u32_k(unsigned* p, unsigned v, size_t n) {
    size_t i = (size_t)blockIdx.x * blockDim.x + threadIdx.x;
    if (i < n) p[i] = v;
}
__global__ void mix_k(const float* __restrict__ a, const float* __restrict__ b,
                      float* __restrict__ out, size_t n) {
    size_t i = (size_t)blockIdx.x * blockDim.x + threadIdx.x;
    if (i < n) out[i] = (1.0f - kSigma) * a[i] + kSigma * b[i];
}
__global__ void relu_k(float* p, size_t n) {
    size_t i = (size_t)blockIdx.x * blockDim.x + threadIdx.x;
    if (i < n) { float v = p[i]; p[i] = v > 0.0f ? v : 0.0f; }
}
__global__ void copy_f32_k(const float* __restrict__ in, float* __restrict__ out, size_t n) {
    size_t i = (size_t)blockIdx.x * blockDim.x + threadIdx.x;
    if (i < n) out[i] = in[i];
}

// ---------------------------------------------------------------------------
// Edge attention (segment softmax over dst + scatter-aggregate)
// ---------------------------------------------------------------------------
__device__ __forceinline__ unsigned f32_order(float f) {
    unsigned u = __float_as_uint(f);
    return (f < 0.0f) ? ~u : (u | 0x80000000u);
}
__device__ __forceinline__ float f32_unorder(unsigned u) {
    return (u & 0x80000000u) ? __uint_as_float(u & 0x7fffffffu) : __uint_as_float(~u);
}

__global__ void edge_logits_k(const float* __restrict__ qn, const float* __restrict__ kn,
                              const float* __restrict__ e, const int* __restrict__ ei,
                              int C, int dh, float inv_sqrt_dh,
                              float* __restrict__ logits, unsigned* __restrict__ mxu)
{
    int idx = blockIdx.x * blockDim.x + threadIdx.x;
    if (idx >= kE * kHeads) return;
    int eid = idx >> 2, hh = idx & 3;
    int src = ei[eid];
    int dst = ei[kE + eid];
    const float4* qp = (const float4*)(qn + (size_t)dst * C + hh * dh);
    const float4* kp = (const float4*)(kn + (size_t)src * C + hh * dh);
    const float4* ep = (const float4*)(e  + (size_t)eid * C + hh * dh);
    if (eid + 64 < kE) __builtin_prefetch(e + (size_t)(eid + 64) * C, 0, 0);
    float acc = 0.0f;
    int n4 = dh >> 2;
    for (int i = 0; i < n4; ++i) {
        float4 q = qp[i], k = kp[i], ev = ep[i];
        acc += q.x * (k.x + ev.x) + q.y * (k.y + ev.y) +
               q.z * (k.z + ev.z) + q.w * (k.w + ev.w);
    }
    float logit = acc * inv_sqrt_dh;
    logits[idx] = logit;
    atomicMax(mxu + (size_t)dst * kHeads + hh, f32_order(logit));
}

__global__ void edge_expsum_k(const float* __restrict__ logits, const unsigned* __restrict__ mxu,
                              const int* __restrict__ ei,
                              float* __restrict__ wexp, float* __restrict__ den)
{
    int idx = blockIdx.x * blockDim.x + threadIdx.x;
    if (idx >= kE * kHeads) return;
    int eid = idx >> 2, hh = idx & 3;
    int dst = ei[kE + eid];
    float m = f32_unorder(mxu[(size_t)dst * kHeads + hh]);
    float w = __expf(logits[idx] - m);
    wexp[idx] = w;
    atomicAdd(den + (size_t)dst * kHeads + hh, w);
}

__global__ void edge_scatter_k(const float* __restrict__ vn, const float* __restrict__ e,
                               const float* __restrict__ wexp, const float* __restrict__ den,
                               const int* __restrict__ ei, int C, int dh,
                               float* __restrict__ node_out)
{
    size_t idx = (size_t)blockIdx.x * blockDim.x + threadIdx.x;
    size_t total = (size_t)kE * C;
    if (idx >= total) return;
    int eid = (int)(idx / C);
    int c   = (int)(idx % C);
    int hh  = c / dh;
    int src = ei[eid];
    int dst = ei[kE + eid];
    float alpha = wexp[(size_t)eid * kHeads + hh] /
                  (den[(size_t)dst * kHeads + hh] + 1e-16f);
    float val = alpha * (vn[(size_t)src * C + c] + e[(size_t)eid * C + c]);
    atomicAdd(node_out + (size_t)dst * C + c, val);
}

// ---------------------------------------------------------------------------
// Heads: softmax over 16 logits per row; Student-t soft assignment (V=1)
// ---------------------------------------------------------------------------
__global__ void softmax16_k(const float* __restrict__ in, float* __restrict__ out, int n) {
    int i = blockIdx.x * blockDim.x + threadIdx.x;
    if (i >= n) return;
    float v[16], mx = -3.0e38f;
    for (int j = 0; j < 16; ++j) { v[j] = in[i * 16 + j]; mx = v[j] > mx ? v[j] : mx; }
    float s = 0.0f;
    for (int j = 0; j < 16; ++j) { v[j] = __expf(v[j] - mx); s += v[j]; }
    float inv = 1.0f / s;
    for (int j = 0; j < 16; ++j) out[i * 16 + j] = v[j] * inv;
}

__global__ void student_t_k(const float* __restrict__ z, const float* __restrict__ mu,
                            float* __restrict__ q, int n) {
    int i = blockIdx.x * blockDim.x + threadIdx.x;
    if (i >= n) return;
    float qs[16], s = 0.0f;
    for (int k = 0; k < 16; ++k) {
        float d2 = 0.0f;
        for (int j = 0; j < 64; ++j) {
            float d = z[(size_t)i * 64 + j] - mu[k * 64 + j];
            d2 += d * d;
        }
        float t = 1.0f / (1.0f + d2);  // V = 1, exponent (V+1)/2 = 1
        qs[k] = t; s += t;
    }
    float inv = 1.0f / s;
    for (int k = 0; k < 16; ++k) q[(size_t)i * 16 + k] = qs[k] * inv;
}

// ---------------------------------------------------------------------------
// Host orchestration
// ---------------------------------------------------------------------------
static inline unsigned cdiv(size_t a, size_t b) { return (unsigned)((a + b - 1) / b); }

extern "C" void kernel_launch(void* const* d_in, const int* in_sizes, int n_in,
                              void* d_out, int out_size, void* d_ws, size_t ws_size,
                              hipStream_t stream)
{
    (void)in_sizes; (void)n_in; (void)out_size; (void)ws_size;

    // --- inputs (setup_inputs insertion order, tuples flattened (w,b)) ---
    const float* x         = (const float*)d_in[0];
    const int*   edge_idx  = (const int*)d_in[1];
    const float* edge_attr = (const float*)d_in[2];
    // d_in[3] edge_to_edge_index: dead (reference slices conv output [:N])
    auto W = [&](int i) { return (const float*)d_in[i]; };
    const float* mu = (const float*)d_in[80];

    // --- outputs: (x_bar, q, predict, z) concatenated ---
    float* out    = (float*)d_out;
    float* o_xbar = out;                       // kN*256
    float* o_q    = o_xbar + (size_t)kN * 256; // kN*16
    float* o_pred = o_q    + (size_t)kN * 16;  // kN*16
    float* o_z    = o_pred + (size_t)kN * 16;  // kN*64

    // --- workspace carve (assumes ws_size >= ~600 MB; dominated by E x 512 e) ---
    char* p = (char*)d_ws;
    auto take = [&](size_t bytes) -> void* {
        void* r = (void*)p; p += (bytes + 255) & ~(size_t)255; return r;
    };
    float* t1  = (float*)take((size_t)kN * 256 * 4);
    float* t2  = (float*)take((size_t)kN * 256 * 4);
    float* t3  = (float*)take((size_t)kN * 512 * 4);
    float* zf  = (float*)take((size_t)kN * 64 * 4);
    float* dd1 = (float*)take((size_t)kN * 512 * 4);
    float* dd2 = (float*)take((size_t)kN * 256 * 4);
    float* dd3 = (float*)take((size_t)kN * 256 * 4);
    float* h   = (float*)take((size_t)kN * 512 * 4);
    float* hn  = (float*)take((size_t)kN * 512 * 4);
    float* hm  = (float*)take((size_t)kN * 512 * 4);
    float* qn  = (float*)take((size_t)kN * 512 * 4);
    float* kn  = (float*)take((size_t)kN * 512 * 4);
    float* vn  = (float*)take((size_t)kN * 512 * 4);
    float* e   = (float*)take((size_t)kE * 512 * 4);
    float* logits = (float*)take((size_t)kE * kHeads * 4);
    float* wexp   = (float*)take((size_t)kE * kHeads * 4);
    unsigned* mxu = (unsigned*)take((size_t)kN * kHeads * 4);
    float* den    = (float*)take((size_t)kN * kHeads * 4);
    __bf16* Abf = (__bf16*)take((size_t)kN * 512 * 2);
    __bf16* Wbf = (__bf16*)take((size_t)512 * 512 * 2);
    __bf16* Ebf = (__bf16*)take((size_t)kE * 64 * 2);

    // --- helpers ---
    auto cast = [&](const float* src, __bf16* dst, size_t n) {
        cast_f32_bf16_k<<<cdiv(n, 256), 256, 0, stream>>>(src, dst, n);
    };
    auto gemm = [&](const __bf16* Ab, int Kd, const float* Wf, const float* bf,
                    float* outp, int M, int Nc, int act) {
        cast(Wf, Wbf, (size_t)Kd * Nc);
        dim3 g(cdiv(Nc, 64), cdiv(M, 128));
        gemm_bf16_wmma<<<g, 256, 0, stream>>>(Ab, Kd, Wbf, Nc, bf, outp, Nc, M, Nc, Kd, act);
    };
    auto linear = [&](const float* Af, int Kd, const float* Wf, const float* bf,
                      float* outp, int M, int Nc, int act) {
        cast(Af, Abf, (size_t)M * Kd);
        gemm(Abf, Kd, Wf, bf, outp, M, Nc, act);
    };

    auto spatial_conv = [&](const float* xn, int C, int layer, float* node_out) {
        const int dh = C / kHeads;
        const int b = 30 + layer * 10;
        // node q/k/v linears: one activation cast, three WMMA GEMMs
        cast(xn, Abf, (size_t)kN * C);
        gemm(Abf, C, W(b + 0), W(b + 1), qn, kN, C, 0);
        gemm(Abf, C, W(b + 2), W(b + 3), kn, kN, C, 0);
        gemm(Abf, C, W(b + 4), W(b + 5), vn, kN, C, 0);
        // edge linear: e = edge_attr @ We + be  (E x 64 -> E x C)
        gemm(Ebf, 64, W(b + 6), W(b + 7), e, kE, C, 0);
        // segment softmax over dst
        fill_u32_k<<<cdiv((size_t)kN * kHeads, 256), 256, 0, stream>>>(mxu, 0u, (size_t)kN * kHeads);
        fill_f32_k<<<cdiv((size_t)kN * kHeads, 256), 256, 0, stream>>>(den, 0.0f, (size_t)kN * kHeads);
        float inv_sqrt_dh = 1.0f / sqrtf((float)dh);
        edge_logits_k<<<cdiv((size_t)kE * kHeads, 256), 256, 0, stream>>>(
            qn, kn, e, edge_idx, C, dh, inv_sqrt_dh, logits, mxu);
        edge_expsum_k<<<cdiv((size_t)kE * kHeads, 256), 256, 0, stream>>>(
            logits, mxu, edge_idx, wexp, den);
        // scatter-aggregate messages to nodes, then ReLU
        fill_f32_k<<<cdiv((size_t)kN * C, 256), 256, 0, stream>>>(node_out, 0.0f, (size_t)kN * C);
        edge_scatter_k<<<cdiv((size_t)kE * C, 256), 256, 0, stream>>>(
            vn, e, wexp, den, edge_idx, C, dh, node_out);
        relu_k<<<cdiv((size_t)kN * C, 256), 256, 0, stream>>>(node_out, (size_t)kN * C);
    };
    auto mix = [&](const float* a, const float* bfx, size_t n) {
        mix_k<<<cdiv(n, 256), 256, 0, stream>>>(a, bfx, hm, n);
    };

    // ---- Autoencoder ----
    linear(x,   256, W(4),  W(5),  t1,     kN, 256, 1);
    linear(t1,  256, W(6),  W(7),  t2,     kN, 256, 1);
    linear(t2,  256, W(8),  W(9),  t3,     kN, 512, 1);
    linear(t3,  512, W(10), W(11), zf,     kN, 64,  0);
    linear(zf,  64,  W(12), W(13), dd1,    kN, 512, 1);
    linear(dd1, 512, W(14), W(15), dd2,    kN, 256, 1);
    linear(dd2, 256, W(16), W(17), dd3,    kN, 256, 1);
    linear(dd3, 256, W(18), W(19), o_xbar, kN, 256, 0);

    // ---- spatial branch with sigma mixing ----
    cast(edge_attr, Ebf, (size_t)kE * 64);  // constant across all conv layers
    linear(x, 256, W(20), W(21), h, kN, 256, 1);
    spatial_conv(h, 256, 0, hn);                                   // h1
    mix(hn, t1, (size_t)kN * 256); linear(hm, 256, W(22), W(23), h, kN, 256, 1);
    spatial_conv(h, 256, 1, hn);                                   // h2
    mix(hn, t2, (size_t)kN * 256); linear(hm, 256, W(24), W(25), h, kN, 512, 1);
    spatial_conv(h, 512, 2, hn);                                   // h3
    mix(hn, t3, (size_t)kN * 512); linear(hm, 512, W(26), W(27), h, kN, 64, 1);
    spatial_conv(h, 64, 3, hn);                                    // h4
    mix(hn, zf, (size_t)kN * 64);  linear(hm, 64,  W(28), W(29), h, kN, 16, 0); // no relu
    spatial_conv(h, 16, 4, hn);                                    // h5

    softmax16_k<<<cdiv(kN, 256), 256, 0, stream>>>(hn, o_pred, kN);
    student_t_k<<<cdiv(kN, 256), 256, 0, stream>>>(zf, mu, o_q, kN);
    copy_f32_k<<<cdiv((size_t)kN * 64, 256), 256, 0, stream>>>(zf, o_z, (size_t)kN * 64);
}
